// PointManifold2_32298154066461
// MI455X (gfx1250) — compile-verified, matrix operand order checked by reading, not run on hardware
//
#include <hip/hip_runtime.h>
#include <math.h>

typedef _Float16 half_t;
typedef half_t v16h __attribute__((ext_vector_type(16)));
typedef float  v8f  __attribute__((ext_vector_type(8)));

#define B_   16
#define N_   1024
#define K_   20
#define LRELU_ 0.2f

__device__ __forceinline__ float lrelu(float x){ return x >= 0.f ? x : LRELU_*x; }
// order-preserving float <-> uint key (atomic max over floats incl. negatives)
__device__ __forceinline__ unsigned fkey(float f){ unsigned i=__float_as_uint(f); return (i & 0x80000000u) ? ~i : (i | 0x80000000u); }
__device__ __forceinline__ float funkey(unsigned k){ return __uint_as_float((k & 0x80000000u) ? (k & 0x7fffffffu) : ~k); }
// inverse fragment map: channel-in-block cl (0..31) -> (lane-half s, half slot h)
// matches ISA 16-bit A/B layout: lane<16 K={0..7,16..23}, lane>=16 K={8..15,24..31}
__device__ __forceinline__ int frag_s(int cl){ return (cl >> 3) & 1; }
__device__ __forceinline__ int frag_h(int cl){ return (cl & 7) + ((cl & 16) ? 8 : 0); }

// ---------------- knn: top-20 of pd = 2*inner - ||q||^2 - ||m||^2 (stable, tie -> lower index) -------------
__global__ void knn_kernel(const float* __restrict__ feat, int bStride, int C, int* __restrict__ idx)
{
  int tid = blockIdx.x * blockDim.x + threadIdx.x;
  if (tid >= B_*N_) return;
  int b = tid / N_, n = tid % N_;
  const float* fb = feat + (size_t)b * bStride;
  float q[128];
  float xxn = 0.f;
  for (int c = 0; c < C; ++c) { float t = fb[c*N_ + n]; q[c] = t; xxn += t*t; }
  float vals[K_]; int inds[K_];
  for (int i = 0; i < K_; ++i) { vals[i] = -3.0e38f; inds[i] = 0; }
  for (int m = 0; m < N_; ++m) {
    float inner = 0.f, xxm = 0.f;
    for (int c = 0; c < C; ++c) { float t = fb[c*N_ + m]; inner += q[c]*t; xxm += t*t; }
    float pd = 2.f*inner - xxn - xxm;
    if (pd > vals[K_-1]) {
      int p = K_-1;
      while (p > 0 && vals[p-1] < pd) { vals[p] = vals[p-1]; inds[p] = inds[p-1]; --p; }
      vals[p] = pd; inds[p] = m;
    }
  }
  int* row = idx + (size_t)tid * K_;
  for (int k = 0; k < K_; ++k) row[k] = inds[k];
}

// ---------------- manifold weight: cosine(normal[first neighbor], normal[k-th neighbor]) -------------------
__global__ void mweight_kernel(const float* __restrict__ in, const int* __restrict__ idx, float* __restrict__ wgt)
{
  int tid = blockIdx.x * blockDim.x + threadIdx.x;
  if (tid >= B_*N_) return;
  int b = tid / N_;
  const float* nb = in + (size_t)b * 6 * N_ + 3 * N_;   // normals = channels 3..5
  const int* row = idx + (size_t)tid * K_;
  int i0 = row[0];
  float cx = nb[0*N_+i0], cy = nb[1*N_+i0], cz = nb[2*N_+i0];
  float nc = sqrtf(cx*cx + cy*cy + cz*cz);
  float* wrow = wgt + (size_t)tid * K_;
  for (int k = 0; k < K_; ++k) {
    int m = row[k];
    float nx = nb[0*N_+m], ny = nb[1*N_+m], nz = nb[2*N_+m];
    float dot = cx*nx + cy*ny + cz*nz;
    float nn = sqrtf(nx*nx + ny*ny + nz*nz);
    wrow[k] = dot / fmaxf(nc*nn, 1e-8f);
  }
}

// --------- manifold-MLP collapse: conv over channel-constant input -> per-channel affine A*w+B + lrelu -----
__global__ void prep_mp_kernel(const float* __restrict__ wmp, const float* __restrict__ bnmp, int C,
                               float* __restrict__ mpA, float* __restrict__ mpB)
{
  int c = threadIdx.x;
  if (c >= C) return;
  float rs = 0.f;
  for (int c2 = 0; c2 < C; ++c2) rs += wmp[c*C + c2];
  float s = bnmp[c] * rsqrtf(bnmp[3*C + c] + 1e-5f);
  mpA[c] = rs * s;
  mpB[c] = bnmp[C + c] - bnmp[2*C + c] * s;
}

// --------- pre-shuffle a weight matrix W[O][Kdim] (f32) into padded f16 WMMA-fragment layout ---------------
// out record for (kb, o, s): 16 contiguous halves = one lane's A-fragment slice (32B aligned)
__global__ void wfrag_kernel(const float* __restrict__ W, int O, int Kdim, int Kpad, half_t* __restrict__ out)
{
  int t = blockIdx.x * blockDim.x + threadIdx.x;
  if (t >= O * Kpad) return;
  int cl  = t & 31;
  int row = t >> 5;
  int o   = row % O;
  int kb  = row / O;
  int kidx = kb * 32 + cl;
  float v = (kidx < Kdim) ? W[(size_t)o * Kdim + kidx] : 0.f;
  out[((size_t)(kb * O + o)) * 32 + frag_s(cl) * 16 + frag_h(cl)] = (half_t)v;
}

// --------- pre-shuffle H (B,512,N f32) into f16 B-fragment layout for conv5 --------------------------------
__global__ void hfrag_kernel(const float* __restrict__ H, half_t* __restrict__ Hf)
{
  size_t t = (size_t)blockIdx.x * blockDim.x + threadIdx.x;
  if (t >= (size_t)B_ * 512 * N_) return;
  int n = (int)(t % N_);
  size_t r = t / N_;
  int c = (int)(r % 512);
  int b = (int)(r / 512);
  int kb = c >> 5, cl = c & 31;
  Hf[((((size_t)b * 16 + kb) * N_ + n) * 2 + frag_s(cl)) * 16 + frag_h(cl)] = (half_t)H[t];
}

// ---------------- fused edge-conv: build [nf|center] in LDS (fragment order), WMMA, BN+lrelu, max over K ---
__global__ void edgeconv_kernel(const float* __restrict__ feat, int bStride, int C, int Cout,
                                const half_t* __restrict__ wf, const float* __restrict__ bn,
                                const float* __restrict__ mpA, const float* __restrict__ mpB,
                                const int* __restrict__ idx, const float* __restrict__ wgt,
                                float* __restrict__ out, int outBStride)
{
  extern __shared__ __attribute__((aligned(32))) char smem[];
  int b = blockIdx.x;
  int n0 = blockIdx.y * 4;             // 4 points per block -> j = 4*20 = 80 = 5 wmma column tiles
  int twoC = 2 * C;
  int Cpad = (twoC + 31) & ~31;
  int nKB = Cpad >> 5;
  half_t* Bld = (half_t*)smem;                                              // [nKB][80][2][16]
  unsigned* outK = (unsigned*)(smem + (size_t)Cpad * 80 * sizeof(half_t));  // [Cout][4]

  const float* fb = feat + (size_t)b * bStride;
  for (int t = threadIdx.x; t < Cpad * 80; t += blockDim.x) {
    int c = t / 80, j = t % 80;
    int n = n0 + j / K_, k = j % K_;
    float v = 0.f;
    if (c < C) {
      size_t r = ((size_t)b * N_ + n) * K_ + k;
      int m = idx[r];
      float mp = lrelu(mpA[c] * wgt[r] + mpB[c]);
      v = (fb[c*N_ + m] - fb[c*N_ + n]) * mp;        // edge feature * manifold weight
    } else if (c < twoC) {
      v = fb[(c - C)*N_ + n];                        // center branch (k-invariant)
    }
    int kb = c >> 5, cl = c & 31;
    Bld[((size_t)(kb * 80 + j)) * 32 + frag_s(cl) * 16 + frag_h(cl)] = (half_t)v;
  }
  for (int t = threadIdx.x; t < Cout * 4; t += blockDim.x) outK[t] = 0u;
  __syncthreads();

  int lane = threadIdx.x & 31;
  int wv = threadIdx.x >> 5;
  int nwv = blockDim.x >> 5;
  int hs = lane >> 4;
  int tiles = (Cout / 16) * 5;
  const v16h* Af = (const v16h*)wf;
  const v16h* Bv = (const v16h*)Bld;
  const float* gam = bn;
  const float* bet = bn + Cout;
  const float* mn  = bn + 2*Cout;
  const float* vr  = bn + 3*Cout;

  for (int t = wv; t < tiles; t += nwv) {
    int ot = t / 5, jt = t % 5;
    int arow = ot*16 + (lane & 15);
    int bcol = jt*16 + (lane & 15);
    v8f acc = {};
    for (int kb = 0; kb < nKB; ++kb) {
      v16h a  = Af[((size_t)kb * Cout + arow) * 2 + hs];
      v16h bb = Bv[(size_t)(kb * 80 + bcol) * 2 + hs];
      acc = __builtin_amdgcn_wmma_f32_16x16x32_f16(false, a, false, bb, (short)0, acc, false, false);
    }
    int obase = ot*16 + ((lane & 16) ? 8 : 0);
    int nl = bcol / K_;                              // local point 0..3
#pragma unroll
    for (int g = 0; g < 8; ++g) {
      int o = obase + g;
      float s = gam[o] * rsqrtf(vr[o] + 1e-5f);
      float y = lrelu((acc[g] - mn[o]) * s + bet[o]);
      atomicMax(&outK[o * 4 + nl], fkey(y));         // max over K via ordered-uint atomic
    }
  }
  __syncthreads();
  for (int t = threadIdx.x; t < Cout * 4; t += blockDim.x) {
    int o = t / 4, nl = t % 4;
    out[(size_t)b * outBStride + (size_t)o * N_ + n0 + nl] = funkey(outK[t]);
  }
}

// -------- w5 conv (1024x512) + BN + lrelu fused with max & mean pooling over N ----------------------------
__global__ void conv5_kernel(const half_t* __restrict__ w5f, const half_t* __restrict__ hf,
                             const float* __restrict__ bn5, float* __restrict__ hcat)
{
  int b = blockIdx.x;
  int lane = threadIdx.x & 31;
  int wv = threadIdx.x >> 5;
  int hs = lane >> 4;
  int tile = blockIdx.y * 8 + wv;                    // 64 o-tiles of 16
  int obase16 = tile * 16;
  const v16h* Af = (const v16h*)w5f;
  const v16h* Hv = (const v16h*)hf;
  float mx[8], sm[8];
#pragma unroll
  for (int g = 0; g < 8; ++g) { mx[g] = -3.0e38f; sm[g] = 0.f; }
  int arow = obase16 + (lane & 15);
  for (int nt = 0; nt < 64; ++nt) {
    int bcol = nt*16 + (lane & 15);
    if (nt + 1 < 64)   // pull next column tile toward the caches (global_prefetch_b8)
      __builtin_prefetch(&Hv[(((size_t)b * 16) * N_ + (bcol + 16)) * 2 + hs], 0, 1);
    v8f acc = {};
    for (int kb = 0; kb < 16; ++kb) {
      v16h a  = Af[((size_t)kb * 1024 + arow) * 2 + hs];
      v16h bb = Hv[(((size_t)b * 16 + kb) * N_ + bcol) * 2 + hs];
      acc = __builtin_amdgcn_wmma_f32_16x16x32_f16(false, a, false, bb, (short)0, acc, false, false);
    }
    int oadd = (lane & 16) ? 8 : 0;
#pragma unroll
    for (int g = 0; g < 8; ++g) {
      int o = obase16 + oadd + g;
      float s = bn5[o] * rsqrtf(bn5[3*1024 + o] + 1e-5f);
      float y = lrelu((acc[g] - bn5[2*1024 + o]) * s + bn5[1024 + o]);
      mx[g] = fmaxf(mx[g], y);
      sm[g] += y;
    }
  }
#pragma unroll
  for (int off = 8; off >= 1; off >>= 1) {
#pragma unroll
    for (int g = 0; g < 8; ++g) {
      mx[g] = fmaxf(mx[g], __shfl_xor(mx[g], off, 32));
      sm[g] += __shfl_xor(sm[g], off, 32);
    }
  }
  if ((lane & 15) == 0) {
    int oadd = (lane & 16) ? 8 : 0;
#pragma unroll
    for (int g = 0; g < 8; ++g) {
      int o = obase16 + oadd + g;
      hcat[(size_t)b * 2048 + o]        = mx[g];
      hcat[(size_t)b * 2048 + 1024 + o] = sm[g] * (1.f / 1024.f);
    }
  }
}

// ---------------- tiny 3-layer MLP head, one block ----------------
__global__ void mlp_kernel(const float* __restrict__ hcat,
                           const float* __restrict__ lin1w, const float* __restrict__ bn6,
                           const float* __restrict__ lin2w, const float* __restrict__ lin2b, const float* __restrict__ bn7,
                           const float* __restrict__ lin3w, const float* __restrict__ lin3b,
                           float* __restrict__ out)
{
  __shared__ float h1[16 * 512];
  __shared__ float h2[16 * 256];
  for (int t = threadIdx.x; t < 16*512; t += blockDim.x) {
    int r = t / 512, o = t % 512;
    float acc = 0.f;
    const float* hr = hcat + (size_t)r * 2048;
    const float* wr = lin1w + (size_t)o * 2048;
    for (int c = 0; c < 2048; ++c) acc += hr[c] * wr[c];
    float s = bn6[o] * rsqrtf(bn6[3*512 + o] + 1e-5f);
    h1[t] = lrelu((acc - bn6[2*512 + o]) * s + bn6[512 + o]);
  }
  __syncthreads();
  for (int t = threadIdx.x; t < 16*256; t += blockDim.x) {
    int r = t / 256, o = t % 256;
    float acc = lin2b[o];
    const float* hr = h1 + r * 512;
    const float* wr = lin2w + (size_t)o * 512;
    for (int c = 0; c < 512; ++c) acc += hr[c] * wr[c];
    float s = bn7[o] * rsqrtf(bn7[3*256 + o] + 1e-5f);
    h2[t] = lrelu((acc - bn7[2*256 + o]) * s + bn7[256 + o]);
  }
  __syncthreads();
  for (int t = threadIdx.x; t < 16*40; t += blockDim.x) {
    int r = t / 40, o = t % 40;
    float acc = lin3b[o];
    const float* hr = h2 + r * 256;
    const float* wr = lin3w + (size_t)o * 256;
    for (int c = 0; c < 256; ++c) acc += hr[c] * wr[c];
    out[t] = acc;
  }
}

extern "C" void kernel_launch(void* const* d_in, const int* in_sizes, int n_in,
                              void* d_out, int out_size, void* d_ws, size_t ws_size,
                              hipStream_t stream)
{
  (void)in_sizes; (void)n_in; (void)out_size; (void)ws_size;
  const float* x = (const float*)d_in[0];
  const float* w_mp[4]  = {(const float*)d_in[1], (const float*)d_in[3], (const float*)d_in[5], (const float*)d_in[7]};
  const float* bn_mp[4] = {(const float*)d_in[2], (const float*)d_in[4], (const float*)d_in[6], (const float*)d_in[8]};
  const float* wc[4]    = {(const float*)d_in[9],  (const float*)d_in[11], (const float*)d_in[13], (const float*)d_in[15]};
  const float* bnc[4]   = {(const float*)d_in[10], (const float*)d_in[12], (const float*)d_in[14], (const float*)d_in[16]};
  const float* w5    = (const float*)d_in[17];
  const float* bn5   = (const float*)d_in[18];
  const float* lin1w = (const float*)d_in[19];
  const float* bn6   = (const float*)d_in[20];
  const float* lin2w = (const float*)d_in[21];
  const float* lin2b = (const float*)d_in[22];
  const float* bn7   = (const float*)d_in[23];
  const float* lin3w = (const float*)d_in[24];
  const float* lin3b = (const float*)d_in[25];

  float* ws = (float*)d_ws;
  const size_t BNK = (size_t)B_ * N_ * K_;
  float* wgt   = ws;                                  // B*N*K
  int*   idx   = (int*)(ws + BNK);                    // B*N*K
  float* H     = ws + 2 * BNK;                        // B*512*N (x1|x2|x3|x4 channel-concat)
  float* hcat  = H + (size_t)B_ * 512 * N_;           // B*2048 (max|mean)
  float* mpbuf = hcat + (size_t)B_ * 2048;            // 4 * 256
  half_t* fragbase = (half_t*)(mpbuf + 1024);         // 32B-aligned (offset is multiple of 32B)

  const int Cin[4]   = {3, 64, 64, 128};
  const int Cout_[4] = {64, 64, 128, 256};
  const int Kpad_[4] = {32, 128, 128, 256};
  const int inOff[4] = {0, 0, 64*N_, 128*N_};
  const int outOff[4]= {0, 64*N_, 128*N_, 256*N_};

  half_t* wf[4];
  size_t off = 0;
  for (int s = 0; s < 4; ++s) { wf[s] = fragbase + off; off += (size_t)Kpad_[s] * Cout_[s]; }
  half_t* w5f = fragbase + off; off += (size_t)512 * 1024;
  half_t* hf  = fragbase + off;                       // B*512*N halves

  dim3 tb(256);
  // pre-shuffle all conv weights into WMMA fragment layout
  for (int s = 0; s < 4; ++s) {
    int tot = Cout_[s] * Kpad_[s];
    wfrag_kernel<<<dim3((tot + 255)/256), tb, 0, stream>>>(wc[s], Cout_[s], 2*Cin[s], Kpad_[s], wf[s]);
  }
  wfrag_kernel<<<dim3((1024*512 + 255)/256), tb, 0, stream>>>(w5, 1024, 512, 512, w5f);

  for (int s = 0; s < 4; ++s) {
    const float* feat; int bStr;
    if (s == 0) { feat = x; bStr = 6*N_; }
    else        { feat = H + inOff[s]; bStr = 512*N_; }
    knn_kernel<<<dim3((B_*N_ + 255)/256), tb, 0, stream>>>(feat, bStr, Cin[s], idx);
    if (s == 0)
      mweight_kernel<<<dim3((B_*N_ + 255)/256), tb, 0, stream>>>(x, idx, wgt);
    prep_mp_kernel<<<1, 128, 0, stream>>>(w_mp[s], bn_mp[s], Cin[s], mpbuf + s*256, mpbuf + s*256 + 128);
    int Cpad = (2*Cin[s] + 31) & ~31;
    size_t smem = (size_t)Cpad * 80 * sizeof(half_t) + (size_t)Cout_[s] * 4 * sizeof(unsigned);
    edgeconv_kernel<<<dim3(B_, N_/4), tb, smem, stream>>>(feat, bStr, Cin[s], Cout_[s],
        wf[s], bnc[s], mpbuf + s*256, mpbuf + s*256 + 128, idx, wgt, H + outOff[s], 512*N_);
  }
  // convert H to f16 fragment layout, then fused conv5 + pooling
  {
    size_t tot = (size_t)B_ * 512 * N_;
    hfrag_kernel<<<dim3((unsigned)((tot + 255)/256)), tb, 0, stream>>>(H, hf);
  }
  conv5_kernel<<<dim3(B_, 8), tb, 0, stream>>>(w5f, hf, bn5, hcat);
  mlp_kernel<<<1, 512, 0, stream>>>(hcat, lin1w, bn6, lin2w, lin2b, bn7, lin3w, lin3b, (float*)d_out);
}
